// SpatialAttention_22926535426130
// MI455X (gfx1250) — compile-verified, matrix operand order checked by reading, not run on hardware
//
#include <hip/hip_runtime.h>
#include <hip/hip_bf16.h>

typedef __attribute__((ext_vector_type(16))) _Float16 v16h;
typedef __attribute__((ext_vector_type(8)))  _Float16 v8h;
typedef __attribute__((ext_vector_type(8)))  float    v8f;

#define NTOT 4096   // W*D*H
#define CCH  32

// ---------------------------------------------------------------------------
// Stage 1a: Q/K/J 3-tap projections -> Qt, Kt, Jt in (N x 8) layout matching
// the reference's row-major reshape: flat = w*2048 + d*128 + h*8 + o,
// r = flat>>12, n = flat & 4095.  q taps along d, k along w, j along h.
// ---------------------------------------------------------------------------
__global__ __launch_bounds__(256) void proj_qkj_kernel(
    const float* __restrict__ x,
    const float* __restrict__ qw, const float* __restrict__ qb,
    const float* __restrict__ qs, const float* __restrict__ qo,
    const float* __restrict__ kw, const float* __restrict__ kb,
    const float* __restrict__ ks, const float* __restrict__ ko,
    const float* __restrict__ jw, const float* __restrict__ jb,
    const float* __restrict__ js, const float* __restrict__ jo,
    float* __restrict__ qt, float* __restrict__ kt, float* __restrict__ jt,
    _Float16* __restrict__ kth)
{
  const int t = blockIdx.x * blockDim.x + threadIdx.x;   // [0, 2*32768)
  const int b    = t >> 15;
  const int flat = t & 32767;
  const int w = flat >> 11, d = (flat >> 7) & 15, h = (flat >> 3) & 15, o = flat & 7;
  const float* xb = x + b * (NTOT * CCH);

  float accq = 0.f, acck = 0.f, accj = 0.f;
  #pragma unroll
  for (int tap = 0; tap < 3; ++tap) {
    const int off  = tap - 1;
    const int wbase = tap * 256 + o;   // weight flat: tap*C*8 + c*8 + o
    if ((unsigned)(d + off) < 16u) {
      const float* xp = xb + ((w * 16 + (d + off)) * 16 + h) * CCH;
      for (int c = 0; c < CCH; ++c) accq += xp[c] * qw[wbase + c * 8];
    }
    if ((unsigned)(w + off) < 16u) {
      const float* xp = xb + (((w + off) * 16 + d) * 16 + h) * CCH;
      for (int c = 0; c < CCH; ++c) acck += xp[c] * kw[wbase + c * 8];
    }
    if ((unsigned)(h + off) < 16u) {
      const float* xp = xb + ((w * 16 + d) * 16 + (h + off)) * CCH;
      for (int c = 0; c < CCH; ++c) accj += xp[c] * jw[wbase + c * 8];
    }
  }
  const float qv = fmaxf(0.f, (accq + qb[o]) * qs[o] + qo[o]);
  const float kv = fmaxf(0.f, (acck + kb[o]) * ks[o] + ko[o]);
  const float jv = fmaxf(0.f, (accj + jb[o]) * js[o] + jo[o]);

  const int n = flat & 4095, r = flat >> 12;
  const int idx = b * (NTOT * 8) + n * 8 + r;
  qt[idx]  = qv;
  kt[idx]  = kv;
  jt[idx]  = jv;
  kth[idx] = (_Float16)kv;
}

// ---------------------------------------------------------------------------
// Stage 1b: V = relu((x @ v_w + v_b)*v_s + v_o), stored C-major f16: Vt[c][n]
// ---------------------------------------------------------------------------
__global__ __launch_bounds__(256) void proj_v_kernel(
    const float* __restrict__ x,
    const float* __restrict__ vw, const float* __restrict__ vb,
    const float* __restrict__ vs, const float* __restrict__ vo,
    _Float16* __restrict__ vth)
{
  const int t = blockIdx.x * blockDim.x + threadIdx.x;   // [0, 2*4096*32)
  const int b = t >> 17;
  const int rem = t & 131071;
  const int n = rem >> 5, c = rem & 31;
  const float* xp = x + (b * NTOT + n) * CCH;
  float a = 0.f;
  for (int ci = 0; ci < CCH; ++ci) a += xp[ci] * vw[ci * CCH + c];
  a = fmaxf(0.f, (a + vb[c]) * vs[c] + vo[c]);
  vth[b * (CCH * NTOT) + c * NTOT + n] = (_Float16)a;
}

// ---------------------------------------------------------------------------
// Stage 1c: S = K @ Jt  (8x8), reduced over n.  One block per batch.
// ---------------------------------------------------------------------------
__global__ __launch_bounds__(256) void compute_s_kernel(
    const float* __restrict__ kt, const float* __restrict__ jt,
    float* __restrict__ S)
{
  __shared__ float red[256];
  const int b = blockIdx.x;
  const int tid = threadIdx.x;
  const int pair = tid >> 2, part = tid & 3;       // 64 (r,r2) pairs x 4 strides
  const int r = pair >> 3, r2 = pair & 7;
  const float* ktb = kt + b * NTOT * 8;
  const float* jtb = jt + b * NTOT * 8;
  float s = 0.f;
  for (int n = part; n < NTOT; n += 4)
    s += ktb[n * 8 + r] * jtb[n * 8 + r2];
  red[tid] = s;
  __syncthreads();
  if (part == 0)
    S[b * 64 + pair] = red[tid] + red[tid + 1] + red[tid + 2] + red[tid + 3];
}

// ---------------------------------------------------------------------------
// Stage 1d: Q2 = Qt @ S  (N x 8), stored f16.
// ---------------------------------------------------------------------------
__global__ __launch_bounds__(256) void compute_q2_kernel(
    const float* __restrict__ qt, const float* __restrict__ S,
    _Float16* __restrict__ q2h)
{
  const int t = blockIdx.x * blockDim.x + threadIdx.x;   // [0, 2*4096)
  const int b = t >> 12, n = t & 4095;
  const float* qr = qt + (b * NTOT + n) * 8;
  const float* Sb = S + b * 64;
  #pragma unroll
  for (int r2 = 0; r2 < 8; ++r2) {
    float a = 0.f;
    #pragma unroll
    for (int r = 0; r < 8; ++r) a += qr[r] * Sb[r * 8 + r2];
    q2h[(b * NTOT + n) * 8 + r2] = (_Float16)a;
  }
}

// ---------------------------------------------------------------------------
// Main: out = gamma * ( sigmoid(Q2 @ K) @ V ) + x, fully fused per 16-row tile.
// One wave per (batch, 16-row tile); loops j over N in chunks of 32.
//   WMMA#1: P16x16 = Q2pad(16x32, K zero-padded past 8) x Kpad(32x16)  [x2]
//   sigmoid -> f16 -> LDS (D-layout -> row-major) -> reload as A fragment
//   WMMA#2: acc(16x16) += P(16x32) x V(32x16)  for both C halves
// Fragment layouts per ISA 7.12.2 (16-bit A 16x32 / mirrored B 32x16):
//   lanes 0-15 hold K in {0..7, 16..23}; lanes 16-31 hold K in {8..15, 24..31}.
// Sigmoid uses v_exp_f32 + v_rcp_f32 (1-ulp) since the result is immediately
// quantized to f16 anyway; avoids the IEEE div_scale/div_fixup sequence.
// ---------------------------------------------------------------------------
#define PSTRIDE 40   // halfs per LDS P row (80B: 16B-aligned, odd-ish banking)

__global__ __launch_bounds__(256) void attn_main_kernel(
    const _Float16* __restrict__ q2h,   // [B][N][8]
    const _Float16* __restrict__ kth,   // [B][N][8]   (Kt: row n = K[:,n])
    const _Float16* __restrict__ vth,   // [B][C][N]   (C-major)
    const float*    __restrict__ x,
    const float*    __restrict__ gamma,
    float*          __restrict__ out)
{
  __shared__ __align__(16) _Float16 lds_p[8][16 * PSTRIDE];

  const int wave = threadIdx.x >> 5;
  const int lane = threadIdx.x & 31;
  const int wid  = blockIdx.x * 8 + wave;     // 512 waves total
  const int b    = wid >> 8;
  const int ibase = (wid & 255) * 16;

  const _Float16* q2b = q2h + b * NTOT * 8;
  const _Float16* ktb = kth + b * NTOT * 8;
  const _Float16* vtb = vth + b * CCH * NTOT;
  const float*    xb  = x   + b * NTOT * CCH;
  float*          ob  = out + b * NTOT * CCH;

  _Float16* P = &lds_p[wave][0];
  const int  halfLane = lane & 15;
  const bool lo = (lane < 16);

  // A fragment for Q2 (loop-invariant): lanes<16 carry row ibase+lane,
  // K-positions 0..7 (= r). Everything else zero-padded.
  v16h aq;
  #pragma unroll
  for (int e = 0; e < 16; ++e) aq[e] = (_Float16)0.f;
  if (lo) {
    v8h q = *(const v8h*)(q2b + (ibase + lane) * 8);
    #pragma unroll
    for (int e = 0; e < 8; ++e) aq[e] = q[e];
  }

  v8f acc0 = {};
  v8f acc1 = {};
  const float g = gamma[0];

  for (int j0 = 0; j0 < NTOT; j0 += 32) {
    // ---- P = sigmoid(Q2 @ K) for two 16-column subtiles ----
    #pragma unroll
    for (int t = 0; t < 2; ++t) {
      v16h bk;
      #pragma unroll
      for (int e = 0; e < 16; ++e) bk[e] = (_Float16)0.f;
      if (lo) {   // lane = output column j; K-positions 0..7 = Kt row (j0+16t+lane)
        v8h kr = *(const v8h*)(ktb + (j0 + t * 16 + lane) * 8);
        #pragma unroll
        for (int e = 0; e < 8; ++e) bk[e] = kr[e];
      }
      v8f p = {};
      p = __builtin_amdgcn_wmma_f32_16x16x32_f16(false, aq, false, bk,
                                                 (short)0, p, false, false);
      // D layout: lane<16 -> (M=v, col=lane); lane>=16 -> (M=8+v, col=lane-16)
      const int col  = t * 16 + halfLane;
      const int rofs = lo ? 0 : 8;
      #pragma unroll
      for (int v = 0; v < 8; ++v) {
        // sigmoid via fast HW transcendentals: exp2-scaled v_exp_f32 + v_rcp_f32
        const float sig = __builtin_amdgcn_rcpf(1.0f + __expf(-p[v]));
        P[(rofs + v) * PSTRIDE + col] = (_Float16)sig;
      }
    }

    asm volatile("s_wait_dscnt 0" ::: "memory");   // order LDS stores -> loads

    // ---- reload P as A fragment (transpose via LDS) ----
    const int jlo = lo ? 0 : 8;
    v8h plo = *(const v8h*)(P + halfLane * PSTRIDE + jlo);
    v8h phi = *(const v8h*)(P + halfLane * PSTRIDE + jlo + 16);
    v16h ap;
    #pragma unroll
    for (int e = 0; e < 8; ++e) { ap[e] = plo[e]; ap[8 + e] = phi[e]; }

    // ---- V B-fragments for both C halves ----
    const _Float16* v0p = vtb + halfLane * NTOT        + j0 + jlo;
    const _Float16* v1p = vtb + (16 + halfLane) * NTOT + j0 + jlo;
    v8h a0 = *(const v8h*)(v0p);      v8h b0 = *(const v8h*)(v0p + 16);
    v8h a1 = *(const v8h*)(v1p);      v8h b1 = *(const v8h*)(v1p + 16);
    v16h bv0, bv1;
    #pragma unroll
    for (int e = 0; e < 8; ++e) {
      bv0[e] = a0[e]; bv0[8 + e] = b0[e];
      bv1[e] = a1[e]; bv1[8 + e] = b1[e];
    }

    acc0 = __builtin_amdgcn_wmma_f32_16x16x32_f16(false, ap, false, bv0,
                                                  (short)0, acc0, false, false);
    acc1 = __builtin_amdgcn_wmma_f32_16x16x32_f16(false, ap, false, bv1,
                                                  (short)0, acc1, false, false);
    asm volatile("" ::: "memory");   // keep this iter's LDS reads before next stores
  }

  // ---- epilogue: out = gamma*acc + x, straight from D layout ----
  const int c0 = halfLane, c1 = 16 + halfLane;
  const int rbase = ibase + (lo ? 0 : 8);
  #pragma unroll
  for (int v = 0; v < 8; ++v) {
    const int row = rbase + v;
    const int i0 = row * CCH + c0;
    const int i1 = row * CCH + c1;
    ob[i0] = g * acc0[v] + xb[i0];
    ob[i1] = g * acc1[v] + xb[i1];
  }
}

// ---------------------------------------------------------------------------
extern "C" void kernel_launch(void* const* d_in, const int* in_sizes, int n_in,
                              void* d_out, int out_size, void* d_ws, size_t ws_size,
                              hipStream_t stream) {
  const float* x     = (const float*)d_in[0];
  const float* gamma = (const float*)d_in[1];
  const float* q_w = (const float*)d_in[2];
  const float* q_b = (const float*)d_in[3];
  const float* q_s = (const float*)d_in[4];
  const float* q_o = (const float*)d_in[5];
  const float* k_w = (const float*)d_in[6];
  const float* k_b = (const float*)d_in[7];
  const float* k_s = (const float*)d_in[8];
  const float* k_o = (const float*)d_in[9];
  const float* j_w = (const float*)d_in[10];
  const float* j_b = (const float*)d_in[11];
  const float* j_s = (const float*)d_in[12];
  const float* j_o = (const float*)d_in[13];
  const float* v_w = (const float*)d_in[14];
  const float* v_b = (const float*)d_in[15];
  const float* v_s = (const float*)d_in[16];
  const float* v_o = (const float*)d_in[17];
  float* out = (float*)d_out;

  // Workspace carve-up (all offsets 256B-aligned)
  char* ws = (char*)d_ws;
  size_t off = 0;
  float*    qt  = (float*)(ws + off);     off += 2ull * NTOT * 8 * 4;   // 256K
  float*    kt  = (float*)(ws + off);     off += 2ull * NTOT * 8 * 4;   // 256K
  float*    jt  = (float*)(ws + off);     off += 2ull * NTOT * 8 * 4;   // 256K
  float*    S   = (float*)(ws + off);     off += 2ull * 64 * 4;         // 512B
  _Float16* kth = (_Float16*)(ws + off);  off += 2ull * NTOT * 8 * 2;   // 128K
  _Float16* q2h = (_Float16*)(ws + off);  off += 2ull * NTOT * 8 * 2;   // 128K
  _Float16* vth = (_Float16*)(ws + off);  off += 2ull * CCH * NTOT * 2; // 512K

  proj_qkj_kernel<<<256, 256, 0, stream>>>(x,
      q_w, q_b, q_s, q_o, k_w, k_b, k_s, k_o, j_w, j_b, j_s, j_o,
      qt, kt, jt, kth);
  proj_v_kernel<<<1024, 256, 0, stream>>>(x, v_w, v_b, v_s, v_o, vth);
  compute_s_kernel<<<2, 256, 0, stream>>>(kt, jt, S);
  compute_q2_kernel<<<32, 256, 0, stream>>>(qt, S, q2h);
  attn_main_kernel<<<64, 256, 0, stream>>>(q2h, kth, vth, x, gamma, out);
}
